// GeneEmbeddingEncoder_73254962201011
// MI455X (gfx1250) — matrix operand order repaired and checked by reading, not compiled, not run on hardware
//
#include <hip/hip_runtime.h>

// ---------------------------------------------------------------------------
// Problem constants (from reference)
// ---------------------------------------------------------------------------
#define B_SZ      4096
#define G_SZ      20000
#define D_EMB     512
#define H_LOW     512
#define N_HIDDEN  1024
#define TOPK      64
#define INV_T     10.0f      // 1 / 0.1
#define LN_EPS    1e-5f

typedef __attribute__((ext_vector_type(16))) __bf16 v16bf;
typedef __attribute__((ext_vector_type(8)))  __bf16 v8bf;
typedef __attribute__((ext_vector_type(8)))  float  v8f;

// ---------------------------------------------------------------------------
// Async global->LDS staging (gfx1250 GLOBAL_LOAD_ASYNC_TO_LDS_B128, ASYNCcnt).
// Builtin prototype (from the round-2 diagnostic):
//   void __builtin_amdgcn_global_load_async_to_lds_b128(v4i AS1*, v4i AS3*,
//                                                       Imm int off, Imm int cpol)
// Pointers are produced CK-style by casting through integers: flat==AS1
// numerically for global; flat->AS3 is "discard upper 32 bits" per the ISA.
// ---------------------------------------------------------------------------
#if defined(__has_builtin)
#if __has_builtin(__builtin_amdgcn_global_load_async_to_lds_b128)
#define HAVE_ASYNC_LDS 1
#endif
#endif

typedef int v4i_ __attribute__((vector_size(16)));
#define AS1Q __attribute__((address_space(1)))
#define AS3Q __attribute__((address_space(3)))

#ifdef HAVE_ASYNC_LDS
#define ASYNC16(gp, lp)                                                      \
  __builtin_amdgcn_global_load_async_to_lds_b128(                            \
      (AS1Q v4i_*)(unsigned long long)(gp),                                  \
      (AS3Q v4i_*)(unsigned)(unsigned long long)(lp), 0, 0)
static __device__ inline void wait_async_all() {
#if __has_builtin(__builtin_amdgcn_s_wait_asynccnt)
  __builtin_amdgcn_s_wait_asynccnt(0);
#else
  asm volatile("s_wait_asynccnt 0x0" ::: "memory");
#endif
}
#else
#define ASYNC16(gp, lp)                                                      \
  do { *(v8bf*)(lp) = *(const v8bf*)(gp); } while (0)
static __device__ inline void wait_async_all() {}
#endif

// ---------------------------------------------------------------------------
// fp32 -> bf16 conversion (grid-stride)
// ---------------------------------------------------------------------------
__global__ __launch_bounds__(256)
void cvt_bf16_kernel(const float* __restrict__ in, __bf16* __restrict__ out,
                     long long n) {
  long long i = (long long)blockIdx.x * blockDim.x + threadIdx.x;
  long long stride = (long long)gridDim.x * blockDim.x;
  for (; i < n; i += stride) out[i] = (__bf16)in[i];
}

// ---------------------------------------------------------------------------
// WMMA GEMM (NT): C[M,N] = alpha * A[M,K] * B[N,K]^T (+ bias[N])
// A, B bf16 row-major with K contiguous. C fp32 row-major.
// WG = 256 threads = 8 waves; WG tile 128x128; wave tile 64x32 (4x2 of 16x16).
// A/B K-tiles (128x32 bf16) staged in LDS via async global->LDS copies,
// double-buffered; padded LDS row stride 40 elems (80B) keeps ds_load_b128
// aligned and the 16-lane fragment reads conflict-free.
// ---------------------------------------------------------------------------
#define LDSW 40
__global__ __launch_bounds__(256)
void wmma_gemm_nt_kernel(const __bf16* __restrict__ A,
                         const __bf16* __restrict__ B,
                         const float* __restrict__ bias,
                         float* __restrict__ C,
                         int M, int N, int K, float alpha) {
  __shared__ __bf16 As[2][128 * LDSW];
  __shared__ __bf16 Bs[2][128 * LDSW];

  const int lane  = threadIdx.x & 31;
  const int wave  = threadIdx.x >> 5;
  const int wm    = wave >> 2;      // 0..1
  const int wn    = wave & 3;       // 0..3
  const int m0    = blockIdx.y * 128 + wm * 64;
  const int n0    = blockIdx.x * 128 + wn * 32;
  const int lhalf = lane >> 4;      // 0/1
  const int lmod  = lane & 15;

  const v8f vzero = {0.f, 0.f, 0.f, 0.f, 0.f, 0.f, 0.f, 0.f};
  v8f acc[4][2];
#pragma unroll
  for (int i = 0; i < 4; i++)
#pragma unroll
    for (int j = 0; j < 2; j++) acc[i][j] = vzero;

  // --- staging assignment: thread t copies 32B of one A row and one B row --
  const int tr   = threadIdx.x >> 1;        // 0..127 (tile row / col)
  const int tseg = (threadIdx.x & 1) * 16;  // element offset 0 / 16
  int aR = blockIdx.y * 128 + tr; aR = aR < M ? aR : M - 1;
  int bC = blockIdx.x * 128 + tr; bC = bC < N ? bC : N - 1;
  const __bf16* gArow = A + (size_t)aR * K + tseg;
  const __bf16* gBrow = B + (size_t)bC * K + tseg;
  __bf16* laBase = &As[0][tr * LDSW + tseg];
  __bf16* lbBase = &Bs[0][tr * LDSW + tseg];
  const int bufStride = 128 * LDSW;

  // prologue: stage k=0 into buffer 0
  {
    const __bf16* ga = gArow;
    const __bf16* gb = gBrow;
    ASYNC16(ga, laBase);  ASYNC16(ga + 8, laBase + 8);
    ASYNC16(gb, lbBase);  ASYNC16(gb + 8, lbBase + 8);
  }

  int buf = 0;
  for (int k = 0; k < K; k += 32) {
    wait_async_all();      // my async copies into current buffer done
    __syncthreads();       // everyone's copies visible; prior reads finished

    if (k + 32 < K) {      // stage next K-tile into the other buffer
      const int nb = buf ^ 1;
      const __bf16* ga = gArow + (k + 32);
      const __bf16* gb = gBrow + (k + 32);
      __bf16* la = laBase + nb * bufStride;
      __bf16* lb = lbBase + nb * bufStride;
      ASYNC16(ga, la);  ASYNC16(ga + 8, la + 8);
      ASYNC16(gb, lb);  ASYNC16(gb + 8, lb + 8);
      if (k + 96 < K)  // gfx1250 global_prefetch_b8 ahead on the A stream
        __builtin_prefetch(gArow + k + 96, 0, 1);
    }

    // --- fragments from LDS ---
    v16bf af[4], bq[2];
#pragma unroll
    for (int t = 0; t < 4; t++) {
      // A 16x32 layout: lane-half 0 -> K[0..7] & K[16..23], half 1 -> +8
      const __bf16* pa =
          &As[0][buf * bufStride + (wm * 64 + t * 16 + lmod) * LDSW + lhalf * 8];
      v8bf lo = *(const v8bf*)pa;
      v8bf hi = *(const v8bf*)(pa + 16);
      af[t] = __builtin_shufflevector(lo, hi, 0, 1, 2, 3, 4, 5, 6, 7,
                                      8, 9, 10, 11, 12, 13, 14, 15);
    }
#pragma unroll
    for (int t = 0; t < 2; t++) {
      // B 32x16 layout: lane-half 0 -> K[0..7] & K[16..23], half 1 -> +8
      const __bf16* pb =
          &Bs[0][buf * bufStride + (wn * 32 + t * 16 + lmod) * LDSW + lhalf * 8];
      v8bf lo = *(const v8bf*)pb;
      v8bf hi = *(const v8bf*)(pb + 16);
      bq[t] = __builtin_shufflevector(lo, hi, 0, 1, 2, 3, 4, 5, 6, 7,
                                      8, 9, 10, 11, 12, 13, 14, 15);
    }
#pragma unroll
    for (int i = 0; i < 4; i++)
#pragma unroll
      for (int j = 0; j < 2; j++)
        acc[i][j] = __builtin_amdgcn_wmma_f32_16x16x32_bf16(
            false, af[i], false, bq[j], (short)0, acc[i][j], false, false);

    buf ^= 1;
  }

  // C 16x16 f32 layout: VGPR r -> row (lhalf ? 8 : 0) + r, col = lmod
#pragma unroll
  for (int i = 0; i < 4; i++) {
#pragma unroll
    for (int j = 0; j < 2; j++) {
      const int colg = n0 + j * 16 + lmod;
#pragma unroll
      for (int r = 0; r < 8; r++) {
        int rowg = m0 + i * 16 + lhalf * 8 + r;
        if (rowg < M && colg < N) {
          float v = acc[i][j][r] * alpha;
          if (bias) v += bias[colg];
          C[(size_t)rowg * N + colg] = v;
        }
      }
    }
  }
}

// ---------------------------------------------------------------------------
// Per-row LayerNorm + exact GELU
// ---------------------------------------------------------------------------
__global__ __launch_bounds__(256)
void ln_gelu_kernel(const float* __restrict__ in, const float* __restrict__ g,
                    const float* __restrict__ b, float* __restrict__ out,
                    int cols) {
  const int row = blockIdx.x;
  const float* x = in + (size_t)row * cols;
  __shared__ float red[256];

  float s = 0.f, s2 = 0.f;
  for (int c = threadIdx.x; c < cols; c += 256) {
    float v = x[c];
    s += v;
    s2 += v * v;
  }
  red[threadIdx.x] = s;
  __syncthreads();
  for (int o = 128; o > 0; o >>= 1) {
    if (threadIdx.x < o) red[threadIdx.x] += red[threadIdx.x + o];
    __syncthreads();
  }
  const float mean = red[0] / (float)cols;
  __syncthreads();
  red[threadIdx.x] = s2;
  __syncthreads();
  for (int o = 128; o > 0; o >>= 1) {
    if (threadIdx.x < o) red[threadIdx.x] += red[threadIdx.x + o];
    __syncthreads();
  }
  const float var = red[0] / (float)cols - mean * mean;
  const float rstd = rsqrtf(var + LN_EPS);

  for (int c = threadIdx.x; c < cols; c += 256) {
    float y = (x[c] - mean) * rstd * g[c] + b[c];
    out[(size_t)row * cols + c] = 0.5f * y * (1.f + erff(y * 0.70710678f));
  }
}

// ---------------------------------------------------------------------------
// Per-row L2 normalize
// ---------------------------------------------------------------------------
__global__ __launch_bounds__(256)
void l2norm_kernel(const float* __restrict__ in, float* __restrict__ out,
                   int cols) {
  const int row = blockIdx.x;
  const float* x = in + (size_t)row * cols;
  __shared__ float red[256];

  float s2 = 0.f;
  for (int c = threadIdx.x; c < cols; c += 256) {
    float v = x[c];
    s2 += v * v;
  }
  red[threadIdx.x] = s2;
  __syncthreads();
  for (int o = 128; o > 0; o >>= 1) {
    if (threadIdx.x < o) red[threadIdx.x] += red[threadIdx.x + o];
    __syncthreads();
  }
  float n = sqrtf(red[0]);
  float inv = 1.f / fmaxf(n, 1e-12f);
  for (int c = threadIdx.x; c < cols; c += 256)
    out[(size_t)row * cols + c] = x[c] * inv;
}

// ---------------------------------------------------------------------------
// Per-row top-64 (histogram radix-select) + masked softmax + sparse ctx gather
// One 256-thread workgroup per row of scores[B,G]; ctx[b,:] = sum w_i * e[idx_i,:]
// ---------------------------------------------------------------------------
#define CAND 512
__global__ __launch_bounds__(256)
void topk_ctx_kernel(const float* __restrict__ scores,
                     const float* __restrict__ e, float* __restrict__ ctx) {
  const int b = blockIdx.x;
  const int tid = threadIdx.x;
  const float* srow = scores + (size_t)b * G_SZ;

  __shared__ float red[256];
  __shared__ int hist[1024];
  __shared__ float cval[CAND];
  __shared__ int cidx[CAND];
  __shared__ float sw[TOPK];
  __shared__ int s_cnt, s_bin, s_kk;
  __shared__ float s_max, s_min, s_thr, s_den;

  // pass 1: row max / min
  float lmax = -3.0e38f, lmin = 3.0e38f;
  for (int j = tid; j < G_SZ; j += 256) {
    float v = srow[j];
    lmax = fmaxf(lmax, v);
    lmin = fminf(lmin, v);
  }
  red[tid] = lmax;
  __syncthreads();
  for (int o = 128; o > 0; o >>= 1) {
    if (tid < o) red[tid] = fmaxf(red[tid], red[tid + o]);
    __syncthreads();
  }
  if (tid == 0) s_max = red[0];
  __syncthreads();
  red[tid] = lmin;
  __syncthreads();
  for (int o = 128; o > 0; o >>= 1) {
    if (tid < o) red[tid] = fminf(red[tid], red[tid + o]);
    __syncthreads();
  }
  if (tid == 0) s_min = red[0];

  // histogram (bin 0 = largest values)
  for (int i = tid; i < 1024; i += 256) hist[i] = 0;
  __syncthreads();
  const float binw = (s_max - s_min) * (1.0f / 1024.0f) + 1e-12f;
  const float invw = 1.0f / binw;
  for (int j = tid; j < G_SZ; j += 256) {
    int bin = (int)((s_max - srow[j]) * invw);
    bin = bin < 0 ? 0 : (bin > 1023 ? 1023 : bin);
    atomicAdd(&hist[bin], 1);
  }
  __syncthreads();
  if (tid == 0) {
    int cum = 0, sel = 1023;
    for (int i = 0; i < 1024; i++) {
      cum += hist[i];
      if (cum >= TOPK) { sel = i; break; }
    }
    s_bin = sel;
    s_thr = s_max - (float)(sel + 1) * binw;  // lower edge; >=64 values above
    s_cnt = 0;
  }
  __syncthreads();

  // collect candidates >= threshold edge
  const float thr = s_thr;
  for (int j = tid; j < G_SZ; j += 256) {
    float v = srow[j];
    if (v >= thr) {
      int p = atomicAdd(&s_cnt, 1);
      if (p < CAND) { cval[p] = v; cidx[p] = j; }
    }
  }
  __syncthreads();

  // exact top-64 among candidates (partial selection sort, thread 0)
  if (tid == 0) {
    int nc = s_cnt < CAND ? s_cnt : CAND;
    int kk = nc < TOPK ? nc : TOPK;
    for (int r = 0; r < kk; r++) {
      int best = r;
      float bv = cval[r];
      for (int t = r + 1; t < nc; t++)
        if (cval[t] > bv) { bv = cval[t]; best = t; }
      float tv = cval[r]; cval[r] = cval[best]; cval[best] = tv;
      int ti = cidx[r];   cidx[r] = cidx[best]; cidx[best] = ti;
    }
    s_kk = kk;
  }
  __syncthreads();

  // softmax over selected
  const int kk = s_kk;
  if (tid < kk) sw[tid] = __expf(cval[tid] - cval[0]);
  __syncthreads();
  if (tid == 0) {
    float d = 0.f;
    for (int i = 0; i < kk; i++) d += sw[i];
    s_den = d;
  }
  __syncthreads();
  const float invd = 1.0f / s_den;

  // ctx[b, :]: each thread owns 2 of the 512 dims
  const int d0 = tid * 2;
  float a0 = 0.f, a1 = 0.f;
  for (int i = 0; i < kk; i++) {
    float w = sw[i] * invd;
    const float* erow = e + (size_t)cidx[i] * H_LOW;
    a0 += w * erow[d0];
    a1 += w * erow[d0 + 1];
  }
  ctx[(size_t)b * H_LOW + d0]     = a0;
  ctx[(size_t)b * H_LOW + d0 + 1] = a1;
}

// ---------------------------------------------------------------------------
// FiLM combine: h_cell = (1 + gamma) * h_x + beta   (FILM_W = 1)
// ---------------------------------------------------------------------------
__global__ __launch_bounds__(256)
void film_kernel(const float* __restrict__ hx, const float* __restrict__ gamma,
                 const float* __restrict__ beta, float* __restrict__ out,
                 long long n) {
  long long i = (long long)blockIdx.x * blockDim.x + threadIdx.x;
  long long stride = (long long)gridDim.x * blockDim.x;
  for (; i < n; i += stride)
    out[i] = (1.0f + gamma[i]) * hx[i] + beta[i];
}

// ---------------------------------------------------------------------------
// Host-side orchestration
// ---------------------------------------------------------------------------
static inline dim3 gemm_grid(int M, int N) {
  return dim3((unsigned)((N + 127) / 128), (unsigned)((M + 127) / 128));
}

extern "C" void kernel_launch(void* const* d_in, const int* in_sizes, int n_in,
                              void* d_out, int out_size, void* d_ws,
                              size_t ws_size, hipStream_t stream) {
  const float* x       = (const float*)d_in[0];
  const float* g       = (const float*)d_in[1];
  const float* W_enc   = (const float*)d_in[2];
  const float* b_enc   = (const float*)d_in[3];
  const float* ln1_g   = (const float*)d_in[4];
  const float* ln1_b   = (const float*)d_in[5];
  const float* W_ga    = (const float*)d_in[6];
  const float* W_gamma = (const float*)d_in[7];
  const float* b_gamma = (const float*)d_in[8];
  const float* W_beta  = (const float*)d_in[9];
  const float* b_beta  = (const float*)d_in[10];
  const float* W_exp   = (const float*)d_in[11];
  const float* b_exp   = (const float*)d_in[12];
  const float* ln2_g   = (const float*)d_in[13];
  const float* ln2_b   = (const float*)d_in[14];
  float* out = (float*)d_out;

  char* ws = (char*)d_ws;
  // Region 1 (327.68 MB): first xb+wencb, later reused for scores
  __bf16* xb     = (__bf16*)(ws + 0);                       // 163,840,000 B
  __bf16* wencb  = (__bf16*)(ws + 163840000);               //  20,480,000 B
  float*  scores = (float*)(ws + 0);                        // 327,680,000 B (aliases xb/wencb)
  size_t o = 327680000;
  float*  h_lin  = (float*)(ws + o); o += 8388608;
  float*  h_act  = (float*)(ws + o); o += 8388608;
  float*  hx     = (float*)(ws + o); o += 8388608;
  __bf16* hxb    = (__bf16*)(ws + o); o += 4194304;
  __bf16* gb     = (__bf16*)(ws + o); o += 20480000;
  __bf16* wgab   = (__bf16*)(ws + o); o += 524288;
  float*  e_lin  = (float*)(ws + o); o += 40960000;
  float*  e_n    = (float*)(ws + o); o += 40960000;
  __bf16* eb     = (__bf16*)(ws + o); o += 20480000;
  float*  ctx    = (float*)(ws + o); o += 8388608;
  __bf16* ctxb   = (__bf16*)(ws + o); o += 4194304;
  __bf16* wgmb   = (__bf16*)(ws + o); o += 524288;
  __bf16* wbtb   = (__bf16*)(ws + o); o += 524288;
  float*  gma    = (float*)(ws + o); o += 8388608;
  float*  bta    = (float*)(ws + o); o += 8388608;
  float*  hcell  = (float*)(ws + o); o += 8388608;
  __bf16* hcellb = (__bf16*)(ws + o); o += 4194304;
  __bf16* wexpb  = (__bf16*)(ws + o); o += 1048576;
  float*  h2     = (float*)(ws + o); o += 16777216;
  (void)ws_size; (void)in_sizes; (void)n_in; (void)out_size;

  const dim3 blk(256);
  const dim3 cvt_grid(8192);

  // --- encoder GEMM: h = x @ W_enc^T + b_enc ------------------------------
  cvt_bf16_kernel<<<cvt_grid, blk, 0, stream>>>(x, xb, (long long)B_SZ * G_SZ);
  cvt_bf16_kernel<<<cvt_grid, blk, 0, stream>>>(W_enc, wencb, (long long)H_LOW * G_SZ);
  wmma_gemm_nt_kernel<<<gemm_grid(B_SZ, H_LOW), blk, 0, stream>>>(
      xb, wencb, b_enc, h_lin, B_SZ, H_LOW, G_SZ, 1.0f);
  ln_gelu_kernel<<<dim3(B_SZ), blk, 0, stream>>>(h_lin, ln1_g, ln1_b, h_act, H_LOW);
  l2norm_kernel<<<dim3(B_SZ), blk, 0, stream>>>(h_act, hx, H_LOW);
  cvt_bf16_kernel<<<cvt_grid, blk, 0, stream>>>(hx, hxb, (long long)B_SZ * H_LOW);

  // --- gene adapter: e = l2norm(g @ W_ga^T) -------------------------------
  cvt_bf16_kernel<<<cvt_grid, blk, 0, stream>>>(g, gb, (long long)G_SZ * D_EMB);
  cvt_bf16_kernel<<<cvt_grid, blk, 0, stream>>>(W_ga, wgab, (long long)H_LOW * D_EMB);
  wmma_gemm_nt_kernel<<<gemm_grid(G_SZ, H_LOW), blk, 0, stream>>>(
      gb, wgab, nullptr, e_lin, G_SZ, H_LOW, D_EMB, 1.0f);
  l2norm_kernel<<<dim3(G_SZ), blk, 0, stream>>>(e_lin, e_n, H_LOW);
  cvt_bf16_kernel<<<cvt_grid, blk, 0, stream>>>(e_n, eb, (long long)G_SZ * H_LOW);

  // --- scores = (h_x @ e^T) / T  (alpha = 10); scores aliases dead xb ------
  wmma_gemm_nt_kernel<<<gemm_grid(B_SZ, G_SZ), blk, 0, stream>>>(
      hxb, eb, nullptr, scores, B_SZ, G_SZ, H_LOW, INV_T);

  // --- top-64 + masked softmax + ctx gather -------------------------------
  topk_ctx_kernel<<<dim3(B_SZ), blk, 0, stream>>>(scores, e_n, ctx);

  // --- FiLM: gamma/beta GEMMs, combine ------------------------------------
  cvt_bf16_kernel<<<cvt_grid, blk, 0, stream>>>(ctx, ctxb, (long long)B_SZ * H_LOW);
  cvt_bf16_kernel<<<cvt_grid, blk, 0, stream>>>(W_gamma, wgmb, (long long)H_LOW * H_LOW);
  cvt_bf16_kernel<<<cvt_grid, blk, 0, stream>>>(W_beta, wbtb, (long long)H_LOW * H_LOW);
  wmma_gemm_nt_kernel<<<gemm_grid(B_SZ, H_LOW), blk, 0, stream>>>(
      ctxb, wgmb, b_gamma, gma, B_SZ, H_LOW, H_LOW, 1.0f);
  wmma_gemm_nt_kernel<<<gemm_grid(B_SZ, H_LOW), blk, 0, stream>>>(
      ctxb, wbtb, b_beta, bta, B_SZ, H_LOW, H_LOW, 1.0f);
  film_kernel<<<cvt_grid, blk, 0, stream>>>(hx, gma, bta, hcell,
                                            (long long)B_SZ * H_LOW);

  // --- expand: h2 = h_cell @ W_exp^T + b_exp; LN + GELU -> out ------------
  cvt_bf16_kernel<<<cvt_grid, blk, 0, stream>>>(hcell, hcellb, (long long)B_SZ * H_LOW);
  cvt_bf16_kernel<<<cvt_grid, blk, 0, stream>>>(W_exp, wexpb, (long long)N_HIDDEN * H_LOW);
  wmma_gemm_nt_kernel<<<gemm_grid(B_SZ, N_HIDDEN), blk, 0, stream>>>(
      hcellb, wexpb, b_exp, h2, B_SZ, N_HIDDEN, H_LOW, 1.0f);
  ln_gelu_kernel<<<dim3(B_SZ), blk, 0, stream>>>(h2, ln2_g, ln2_b, out, N_HIDDEN);
}
